// FFQNet_81338090651771
// MI455X (gfx1250) — compile-verified, matrix operand order checked by reading, not run on hardware
//
#include <hip/hip_runtime.h>
#include <hip/hip_bf16.h>

// ---------------------------------------------------------------------------
// FFQNet Q-values for all 64 actions, restructured:
//   G = states @ W1[:,:512]^T          (computed ONCE: 4.3 GFLOP)
//   h1(a,b) = relu((G[b] + W1[:,512+a]) * rn1[b] + b1)   (rn1 action-indep.)
//   h2 = relu(W2 @ norm(h1) + b2); q = Wq.h2 + bq        (275 GFLOP, dominant)
// fused2 v3: A (h1) register-resident in WMMA fragment layout (256 VGPRs),
// row-norm factored out of the GEMM; W2 staged per-N-tile in LDS shared by
// 8 M-tiles (W2 L2 traffic 16.8GB -> 2.1GB), staging software-pipelined
// through 32 prefetch VGPRs so global latency overlaps the WMMA loop.
// Workspace ~23.5 MB.
// ---------------------------------------------------------------------------

typedef __attribute__((ext_vector_type(16))) _Float16 v16h;
typedef __attribute__((ext_vector_type(8)))  _Float16 v8h;
typedef __attribute__((ext_vector_type(8)))  float    v8f;
typedef __attribute__((ext_vector_type(4)))  float    v4f;

#define B_N   4096
#define S_DIM 512
#define A_N   64
#define H1_N  1024
#define H2_N  512
#define EPSF  1e-8f

// --------------------------- prep: states -> f16 + rn1 ----------------------
__global__ __launch_bounds__(256) void prep_states(const float* __restrict__ st,
                                                   _Float16* __restrict__ stH,
                                                   float* __restrict__ rn1) {
  const int wave = threadIdx.x >> 5, lane = threadIdx.x & 31;
  const int b = blockIdx.x * 8 + wave;
  const float* row = st + (size_t)b * S_DIM;
  _Float16* rowH   = stH + (size_t)b * S_DIM;
  float ss = 0.f;
#pragma unroll
  for (int i = 0; i < 4; ++i) {
    const int j = i * 128 + lane * 4;
    v4f x = *(const v4f*)(row + j);
    ss += x[0]*x[0] + x[1]*x[1] + x[2]*x[2] + x[3]*x[3];
    rowH[j+0] = (_Float16)x[0]; rowH[j+1] = (_Float16)x[1];
    rowH[j+2] = (_Float16)x[2]; rowH[j+3] = (_Float16)x[3];
  }
#pragma unroll
  for (int m = 16; m >= 1; m >>= 1) ss += __shfl_xor(ss, m, 32);
  if (lane == 0) rn1[b] = 1.f / (sqrtf(ss + 1.f) + EPSF);   // ||x||=sqrt(||s||^2+1)
}

// ------------------- prep: weight conversions / extraction ------------------
__global__ __launch_bounds__(256) void prep_w1(const float* __restrict__ W1,
                                               _Float16* __restrict__ w1H) {
  const int idx = blockIdx.x * 256 + threadIdx.x;      // over H1*S
  const int j = idx >> 9, k = idx & 511;
  w1H[idx] = (_Float16)W1[(size_t)j * (S_DIM + A_N) + k];
}

__global__ __launch_bounds__(256) void prep_w1a(const float* __restrict__ W1,
                                                float* __restrict__ w1a) {
  const int idx = blockIdx.x * 256 + threadIdx.x;      // over A*H1
  const int act = idx >> 10, j = idx & 1023;
  w1a[act * H1_N + j] = W1[(size_t)j * (S_DIM + A_N) + S_DIM + act];
}

__global__ __launch_bounds__(256) void prep_w2(const float* __restrict__ W2,
                                               _Float16* __restrict__ w2H) {
  const int idx = blockIdx.x * 256 + threadIdx.x;      // over H2*H1
  w2H[idx] = (_Float16)W2[idx];
}

// ------------------------- GEMM1: G = stH @ w1H^T ---------------------------
__global__ __launch_bounds__(256) void gemm1(const _Float16* __restrict__ stH,
                                             const _Float16* __restrict__ w1H,
                                             float* __restrict__ G) {
  const int wave = threadIdx.x >> 5, lane = threadIdx.x & 31;
  const int tile = blockIdx.x * 8 + wave;              // 256*64 tiles
  const int nt = tile & 63, mt = tile >> 6;
  const int m0 = mt * 16, n0 = nt * 16;
  const int mr = lane & 15, g = lane >> 4;

  const _Float16* arow = stH + (size_t)(m0 + mr) * S_DIM;  // A: M = lane&15
  const _Float16* brow = w1H + (size_t)(n0 + mr) * S_DIM;  // B: N = lane&15
  v8f c = {};
#pragma unroll
  for (int k0 = 0; k0 < S_DIM; k0 += 32) {
    v8h alo = *(const v8h*)(arow + k0 + g * 8);
    v8h ahi = *(const v8h*)(arow + k0 + 16 + g * 8);
    v8h blo = *(const v8h*)(brow + k0 + g * 8);
    v8h bhi = *(const v8h*)(brow + k0 + 16 + g * 8);
    v16h av = __builtin_shufflevector(alo, ahi, 0,1,2,3,4,5,6,7,8,9,10,11,12,13,14,15);
    v16h bv = __builtin_shufflevector(blo, bhi, 0,1,2,3,4,5,6,7,8,9,10,11,12,13,14,15);
    c = __builtin_amdgcn_wmma_f32_16x16x32_f16(false, av, false, bv, (short)0, c, false, false);
  }
#pragma unroll
  for (int v = 0; v < 8; ++v)                            // C: VGPR v -> M=v+8g
    G[(size_t)(m0 + v + 8 * g) * H1_N + (n0 + mr)] = c[v];
}

// ---- fused2 v3: reg-resident h1 frags + pipelined LDS-staged W2 ------------
// Block = 256 thr (8 waves) = 1 action x 128 batch rows; wave w owns rows
// [b0+16w, b0+16w+16). h1 for the whole K=1024 lives in v16h afrag[32].
__global__ __launch_bounds__(256) void fused2(const float* __restrict__ G,
                                              const float* __restrict__ w1a,
                                              const float* __restrict__ b1,
                                              const _Float16* __restrict__ w2H,
                                              const float* __restrict__ b2,
                                              const float* __restrict__ Wq,
                                              const float* __restrict__ bq,
                                              const float* __restrict__ rn1,
                                              float* __restrict__ out) {
  __shared__ __align__(16) _Float16 bst[16][H1_N + 8];   // 33 KB, pad-8 halves

  const int act = blockIdx.x >> 5;                       // 32 batch blocks/action
  const int b0  = (blockIdx.x & 31) * 128;
  const int t = threadIdx.x;
  const int wave = t >> 5, lane = t & 31;
  const int mr = lane & 15, g = lane >> 4;

  // ---- layer 1 straight into WMMA A-fragment registers --------------------
  const int b = b0 + wave * 16 + mr;                     // this lane's row
  const float rb = rn1[b];
  const float* Grow = G + (size_t)b * H1_N;
  const float* wa   = w1a + act * H1_N;

  v16h afrag[32];                                        // 16x1024 f16 tile
  float ss = 0.f;
#pragma unroll
  for (int s = 0; s < 32; ++s) {
#pragma unroll
    for (int half = 0; half < 2; ++half) {
      const int kb = s * 32 + half * 16 + g * 8;         // frag k-chunk base
      v4f g0  = *(const v4f*)(Grow + kb);
      v4f g1  = *(const v4f*)(Grow + kb + 4);
      v4f wa0 = *(const v4f*)(wa + kb);
      v4f wa1 = *(const v4f*)(wa + kb + 4);
      v4f c0  = *(const v4f*)(b1 + kb);
      v4f c1  = *(const v4f*)(b1 + kb + 4);
#pragma unroll
      for (int u = 0; u < 4; ++u) {
        float h0 = fmaxf((g0[u] + wa0[u]) * rb + c0[u], 0.f);
        float h1v = fmaxf((g1[u] + wa1[u]) * rb + c1[u], 0.f);
        ss += h0 * h0 + h1v * h1v;
        afrag[s][half * 8 + u]     = (_Float16)h0;       // e<8 : k=kb0+e
        afrag[s][half * 8 + 4 + u] = (_Float16)h1v;      // e>=8: k=kb0+8+e
      }
    }
  }
  // Row sumsq: lanes m and m+16 hold complementary K-halves of row m.
  ss += __shfl_xor(ss, 16, 32);
  const float rn2 = 1.f / (sqrtf(ss) + EPSF);            // norm factored out
  float rn2v[8];
#pragma unroll
  for (int v = 0; v < 8; ++v) rn2v[v] = __shfl(rn2, v + 8 * g, 32); // row v+8g

  float qreg[8] = {0.f, 0.f, 0.f, 0.f, 0.f, 0.f, 0.f, 0.f};

  // This thread's staging chunks: 8 x 16B of a 16x1024 W2 slice.
  int srow[8], scol[8];
#pragma unroll
  for (int i = 0; i < 8; ++i) {
    const int ch = t + i * 256;                          // 2048 chunks
    srow[i] = ch >> 7; scol[i] = (ch & 127) * 8;
  }

  // Prefetch N-tile 0 into registers (32 VGPRs).
  v8h pf[8];
#pragma unroll
  for (int i = 0; i < 8; ++i)
    pf[i] = *(const v8h*)(w2H + (size_t)srow[i] * H1_N + scol[i]);

  // ---- layer 2: 32 N-tiles; staging pipelined through pf registers --------
  for (int nt = 0; nt < 32; ++nt) {
    __syncthreads();                                     // all done reading bst
#pragma unroll
    for (int i = 0; i < 8; ++i)                          // regs -> LDS
      *(v8h*)(&bst[srow[i]][scol[i]]) = pf[i];
    __syncthreads();

    if (nt < 31) {                                       // prefetch next tile;
      const _Float16* src = w2H + (size_t)(nt + 1) * 16 * H1_N;  // overlaps WMMAs
#pragma unroll
      for (int i = 0; i < 8; ++i)
        pf[i] = *(const v8h*)(src + (size_t)srow[i] * H1_N + scol[i]);
    }

    v8f c = {};
    const _Float16* brow = &bst[mr][0];                  // B: N = lane&15
#pragma unroll
    for (int s = 0; s < 32; ++s) {
      const int kb0 = s * 32 + g * 8;
      v8h blo = *(const v8h*)(brow + kb0);
      v8h bhi = *(const v8h*)(brow + kb0 + 16);
      v16h bv = __builtin_shufflevector(blo, bhi, 0,1,2,3,4,5,6,7,8,9,10,11,12,13,14,15);
      c = __builtin_amdgcn_wmma_f32_16x16x32_f16(false, afrag[s], false, bv, (short)0, c, false, false);
    }

    // epilogue: h2 = relu(rn2[m]*C + b2[j]); q += h2 * Wq[j]
    const int j = nt * 16 + mr;
    const float b2j = b2[j], wqj = Wq[j];
#pragma unroll
    for (int v = 0; v < 8; ++v)
      qreg[v] += fmaxf(c[v] * rn2v[v] + b2j, 0.f) * wqj;
  }

  // ---- reduce q over the 16 j-lanes; rows m = v+8g owned by this wave -----
#pragma unroll
  for (int v = 0; v < 8; ++v) {
#pragma unroll
    for (int m = 8; m >= 1; m >>= 1) qreg[v] += __shfl_xor(qreg[v], m, 16);
  }
  if (mr == 0) {
    const float bq0 = bq[0];
#pragma unroll
    for (int v = 0; v < 8; ++v)
      out[(size_t)(b0 + wave * 16 + v + 8 * g) * A_N + act] = qreg[v] + bq0;
  }
}

// ---------------------------------------------------------------------------
extern "C" void kernel_launch(void* const* d_in, const int* in_sizes, int n_in,
                              void* d_out, int out_size, void* d_ws, size_t ws_size,
                              hipStream_t stream) {
  const float* states = (const float*)d_in[0];
  const float* W1     = (const float*)d_in[1];
  const float* b1     = (const float*)d_in[2];
  const float* W2     = (const float*)d_in[3];
  const float* b2     = (const float*)d_in[4];
  const float* Wq     = (const float*)d_in[5];
  const float* bq     = (const float*)d_in[6];
  float* out = (float*)d_out;

  char* ws = (char*)d_ws;                                // ~23.5 MB total
  _Float16* stH = (_Float16*)ws; ws += (size_t)B_N * S_DIM * 2;
  _Float16* w1H = (_Float16*)ws; ws += (size_t)H1_N * S_DIM * 2;
  _Float16* w2H = (_Float16*)ws; ws += (size_t)H2_N * H1_N * 2;
  float*    w1a = (float*)ws;    ws += (size_t)A_N * H1_N * 4;
  float*    rn1 = (float*)ws;    ws += (size_t)B_N * 4;
  float*    G   = (float*)ws;    // B_N*H1_N*4 = 16 MB

  prep_states<<<B_N / 8, 256, 0, stream>>>(states, stH, rn1);
  prep_w1 <<<(H1_N * S_DIM) / 256, 256, 0, stream>>>(W1, w1H);
  prep_w1a<<<(A_N * H1_N) / 256, 256, 0, stream>>>(W1, w1a);
  prep_w2 <<<(H2_N * H1_N) / 256, 256, 0, stream>>>(W2, w2H);
  gemm1   <<<(B_N / 16) * (H1_N / 16) / 8, 256, 0, stream>>>(stH, w1H, G);
  fused2  <<<A_N * (B_N / 128), 256, 0, stream>>>(G, w1a, b1, w2H, b2, Wq, bq, rn1, out);
}
